// CrossAttention_27573690041000
// MI455X (gfx1250) — matrix operand order repaired and checked by reading, not compiled, not run on hardware
//
#include <hip/hip_runtime.h>
#include <hip/hip_bf16.h>
#include <math.h>

// ---------------------------------------------------------------------------
// CDNA5 (gfx1250) cross-attention block, bf16 WMMA GEMMs + f32 softmax/LN.
// 128x64 block tile, 8 waves, wave = 32x32 (4 wmma / K-step).
// Double-buffered LDS tiles filled with GLOBAL_LOAD_ASYNC_TO_LDS_B128
// (ASYNCcnt), overlapping the next tile's DMA with the current WMMAs.
// ---------------------------------------------------------------------------

typedef __attribute__((ext_vector_type(16))) __bf16 v16bf;
typedef __attribute__((ext_vector_type(8)))  float  v8f;

#define BB 16
#define SS 1024
#define EE 1024
#define HH 2048

__device__ __forceinline__ unsigned short f2bf(float x) {
  unsigned int u = __float_as_uint(x);
  unsigned int r = u + 0x7FFFu + ((u >> 16) & 1u);  // round-to-nearest-even
  return (unsigned short)(r >> 16);
}

// async DMA: 16 bytes global -> LDS, tracked by ASYNCcnt (per-lane addresses)
__device__ __forceinline__ void async_b128(unsigned lds_byte_addr, const void* gsrc) {
  asm volatile("global_load_async_to_lds_b128 %0, %1, off"
               :: "v"(lds_byte_addr), "v"(gsrc)
               : "memory");
}
__device__ __forceinline__ unsigned lds_addr(const void* p) {
  return (unsigned)(unsigned long long)p;  // low 32 bits of generic ptr = LDS offset
}

// ---------------- f32 -> bf16 conversion ----------------
__global__ __launch_bounds__(256) void k_f32_to_bf16(const float* __restrict__ x,
                                                     unsigned short* __restrict__ y,
                                                     long long n) {
  long long i = (long long)blockIdx.x * blockDim.x + threadIdx.x;
  if (i < n) y[i] = f2bf(x[i]);
}

// f32 [K][N] -> bf16 [N][K] (transpose through LDS, 32x32 tiles)
__global__ __launch_bounds__(256) void k_f32_to_bf16_T(const float* __restrict__ x,
                                                       unsigned short* __restrict__ y,
                                                       int K, int N) {
  __shared__ unsigned short t[32][33];
  int k0 = blockIdx.y * 32, n0 = blockIdx.x * 32;
  int tx = threadIdx.x & 31, ty = threadIdx.x >> 5;
#pragma unroll
  for (int i = ty; i < 32; i += 8)
    t[i][tx] = f2bf(x[(size_t)(k0 + i) * N + n0 + tx]);
  __syncthreads();
#pragma unroll
  for (int i = ty; i < 32; i += 8)
    y[(size_t)(n0 + i) * K + k0 + tx] = t[tx][i];
}

// ---------------- tiled bf16 WMMA GEMM ----------------
// C[M,N] = alpha * A[M,K] * B.   A row-major bf16.
// OPB==1: B is [N,K] row-major (ldb in K units), i.e. C = A*B^T.
// OPB==0: B is [K,N] row-major.
// Block tile M=128, N=64, K-step 32. 8 waves (4 in M x 2 in N), wave = 32x32.
template <int OPB, int CBF16>
__global__ __launch_bounds__(256) void k_gemm_bf16_wmma(
    const unsigned short* __restrict__ A, const unsigned short* __restrict__ B,
    void* __restrict__ Cv, int K, int lda, int ldb, int ldc,
    long long sA, long long sB, long long sC, float alpha) {
  __shared__ unsigned short aT[2][128 * 40];  // [128 rows][32 K], stride 40
  __shared__ unsigned short bT[2][64 * 40];   // OPB1: [n][k] s40 ; OPB0: [k][n] s72

  const int tid  = threadIdx.x;
  const int lane = tid & 31;
  const int wid  = tid >> 5;
  const int wm   = (wid & 3) * 32;   // wave M offset (4 waves -> 128)
  const int wn   = (wid >> 2) * 32;  // wave N offset (2 waves -> 64)
  const int bm   = blockIdx.y * 128;
  const int bn   = blockIdx.x * 64;

  const unsigned short* Ab = A + sA * blockIdx.z;
  const unsigned short* Bb = B + sB * blockIdx.z;

  const int hf = lane >> 4;   // half-wave select
  const int mn = lane & 15;

  // per-thread staging coordinates
  const int arow = tid >> 2, aseg = (tid & 3) * 8;
  const int brow = OPB ? (tid >> 2) : (tid >> 3);
  const int bseg = OPB ? (tid & 3) * 8 : (tid & 7) * 8;

  // k0 is an ELEMENT offset (multiple of 32)
  auto issue = [&](int k0, int buf) {
    async_b128(lds_addr(&aT[buf][arow * 40 + aseg]),
               Ab + (size_t)(bm + arow) * lda + (k0 + aseg));
    async_b128(lds_addr(&aT[buf][(64 + arow) * 40 + aseg]),
               Ab + (size_t)(bm + 64 + arow) * lda + (k0 + aseg));
    if (OPB) {
      async_b128(lds_addr(&bT[buf][brow * 40 + bseg]),
                 Bb + (size_t)(bn + brow) * ldb + (k0 + bseg));
    } else {
      async_b128(lds_addr(&bT[buf][brow * 72 + bseg]),
                 Bb + (size_t)(k0 + brow) * ldb + (bn + bseg));
    }
  };

  v8f c00 = {}, c01 = {}, c10 = {}, c11 = {};

  auto compute = [&](int buf) {
    const unsigned short* at = aT[buf];
    const unsigned short* bt = bT[buf];

    // A fragments (16x32): VGPR r -> K = hf*16 + 2r
    union { unsigned int u[8]; v16bf v; } fa0, fa1, fb0, fb1;
#pragma unroll
    for (int r = 0; r < 8; ++r) {
      int kk = hf * 16 + r * 2;
      fa0.u[r] = *(const unsigned int*)&at[(wm + mn) * 40 + kk];
      fa1.u[r] = *(const unsigned int*)&at[(wm + 16 + mn) * 40 + kk];
    }
    // B fragments (32x16): VGPR r -> K = hf*16 + 2r, N = mn
    if (OPB) {
#pragma unroll
      for (int r = 0; r < 8; ++r) {
        int kk = hf * 16 + r * 2;
        fb0.u[r] = *(const unsigned int*)&bt[(wn + mn) * 40 + kk];
        fb1.u[r] = *(const unsigned int*)&bt[(wn + 16 + mn) * 40 + kk];
      }
    } else {
#pragma unroll
      for (int r = 0; r < 8; ++r) {
        int kk = hf * 16 + r * 2;
        unsigned int lo0 = bt[kk * 72 + wn + mn];
        unsigned int hi0 = bt[(kk + 1) * 72 + wn + mn];
        fb0.u[r] = lo0 | (hi0 << 16);
        unsigned int lo1 = bt[kk * 72 + wn + 16 + mn];
        unsigned int hi1 = bt[(kk + 1) * 72 + wn + 16 + mn];
        fb1.u[r] = lo1 | (hi1 << 16);
      }
    }
    c00 = __builtin_amdgcn_wmma_f32_16x16x32_bf16(false, fa0.v, false, fb0.v,
                                                  (short)0, c00, false, false);
    c01 = __builtin_amdgcn_wmma_f32_16x16x32_bf16(false, fa0.v, false, fb1.v,
                                                  (short)0, c01, false, false);
    c10 = __builtin_amdgcn_wmma_f32_16x16x32_bf16(false, fa1.v, false, fb0.v,
                                                  (short)0, c10, false, false);
    c11 = __builtin_amdgcn_wmma_f32_16x16x32_bf16(false, fa1.v, false, fb1.v,
                                                  (short)0, c11, false, false);
  };

  const int nk = K >> 5;
  issue(0, 0);

  // steady state: overlap tile t+1's DMA with tile t's WMMAs
  for (int t = 0; t < nk - 1; ++t) {
    const int buf = t & 1;
    issue((t + 1) << 5, buf ^ 1);
    asm volatile("s_wait_asynccnt 0x3" ::: "memory");  // this tile's 3 ops done
    __syncthreads();
    compute(buf);
    __syncthreads();  // done reading buf before it is refilled at t+2
  }
  // final tile
  asm volatile("s_wait_asynccnt 0x0" ::: "memory");
  __syncthreads();
  compute((nk - 1) & 1);

  // write out: C/D layout -> lane(hf,mn), VGPR r holds row hf*8 + r
  long long base = sC * blockIdx.z;
#pragma unroll
  for (int r = 0; r < 8; ++r) {
    int gm0 = bm + wm + hf * 8 + r;
    int gm1 = gm0 + 16;
    int gn0 = bn + wn + mn;
    int gn1 = gn0 + 16;
    float y00 = c00[r] * alpha, y01 = c01[r] * alpha;
    float y10 = c10[r] * alpha, y11 = c11[r] * alpha;
    if (CBF16) {
      unsigned short* C = (unsigned short*)Cv;
      C[base + (long long)gm0 * ldc + gn0] = f2bf(y00);
      C[base + (long long)gm0 * ldc + gn1] = f2bf(y01);
      C[base + (long long)gm1 * ldc + gn0] = f2bf(y10);
      C[base + (long long)gm1 * ldc + gn1] = f2bf(y11);
    } else {
      float* C = (float*)Cv;
      C[base + (long long)gm0 * ldc + gn0] = y00;
      C[base + (long long)gm0 * ldc + gn1] = y01;
      C[base + (long long)gm1 * ldc + gn0] = y10;
      C[base + (long long)gm1 * ldc + gn1] = y11;
    }
  }
}

// ---------------- flattened-per-batch softmax ----------------
__global__ __launch_bounds__(256) void k_part_max(const float* __restrict__ x,
                                                  float* __restrict__ pmax,
                                                  int per_batch, int chunk) {
  __shared__ float sm[256];
  int b = blockIdx.y;
  const float* p = x + (long long)b * per_batch + (long long)blockIdx.x * chunk;
  float m = -INFINITY;
  for (int i = threadIdx.x; i < chunk; i += 256) m = fmaxf(m, p[i]);
  sm[threadIdx.x] = m; __syncthreads();
  for (int s = 128; s > 0; s >>= 1) {
    if (threadIdx.x < s) sm[threadIdx.x] = fmaxf(sm[threadIdx.x], sm[threadIdx.x + s]);
    __syncthreads();
  }
  if (threadIdx.x == 0) pmax[(long long)b * gridDim.x + blockIdx.x] = sm[0];
}

__global__ __launch_bounds__(256) void k_part_expsum(const float* __restrict__ x,
                                                     const float* __restrict__ gmax,
                                                     float* __restrict__ psum,
                                                     int per_batch, int chunk) {
  __shared__ float sm[256];
  int b = blockIdx.y;
  float m = gmax[b];
  const float* p = x + (long long)b * per_batch + (long long)blockIdx.x * chunk;
  float s = 0.f;
  for (int i = threadIdx.x; i < chunk; i += 256) s += expf(p[i] - m);
  sm[threadIdx.x] = s; __syncthreads();
  for (int st = 128; st > 0; st >>= 1) {
    if (threadIdx.x < st) sm[threadIdx.x] += sm[threadIdx.x + st];
    __syncthreads();
  }
  if (threadIdx.x == 0) psum[(long long)b * gridDim.x + blockIdx.x] = sm[0];
}

__global__ __launch_bounds__(256) void k_reduce_final(const float* __restrict__ part,
                                                      float* __restrict__ out,
                                                      int n, int do_max) {
  __shared__ float sm[256];
  int b = blockIdx.x;
  const float* p = part + (long long)b * n;
  float acc = do_max ? -INFINITY : 0.f;
  for (int i = threadIdx.x; i < n; i += 256)
    acc = do_max ? fmaxf(acc, p[i]) : (acc + p[i]);
  sm[threadIdx.x] = acc; __syncthreads();
  for (int s = 128; s > 0; s >>= 1) {
    if (threadIdx.x < s)
      sm[threadIdx.x] = do_max ? fmaxf(sm[threadIdx.x], sm[threadIdx.x + s])
                               : (sm[threadIdx.x] + sm[threadIdx.x + s]);
    __syncthreads();
  }
  if (threadIdx.x == 0) out[b] = sm[0];
}

__global__ __launch_bounds__(256) void k_softmax_norm(const float* __restrict__ x,
                                                      const float* __restrict__ gmax,
                                                      const float* __restrict__ gsum,
                                                      unsigned short* __restrict__ w,
                                                      int per_batch) {
  int b = blockIdx.y;
  long long i = (long long)blockIdx.x * blockDim.x + threadIdx.x;
  if (i < per_batch) {
    float inv = 1.f / gsum[b];
    long long idx = (long long)b * per_batch + i;
    w[idx] = f2bf(expf(x[idx] - gmax[b]) * inv);
  }
}

// ---------------- fused residual + LayerNorm ----------------
__global__ __launch_bounds__(256) void k_ln_fused(const float* __restrict__ a,
                                                  const float* __restrict__ res,
                                                  const float* __restrict__ g,
                                                  const float* __restrict__ bta,
                                                  float* __restrict__ of,
                                                  unsigned short* __restrict__ ob,
                                                  int E) {
  __shared__ float s1[256], s2[256];
  int row = blockIdx.x, tid = threadIdx.x;
  const float* pa = a + (long long)row * E;
  const float* pr = res + (long long)row * E;
  float s = 0.f, q = 0.f;
  for (int i = tid; i < E; i += 256) { float v = pa[i] + pr[i]; s += v; q += v * v; }
  s1[tid] = s; s2[tid] = q; __syncthreads();
  for (int st = 128; st > 0; st >>= 1) {
    if (tid < st) { s1[tid] += s1[tid + st]; s2[tid] += s2[tid + st]; }
    __syncthreads();
  }
  float mu = s1[0] / E;
  float var = s2[0] / E - mu * mu;
  float rs = rsqrtf(var + 1e-5f);
  for (int i = tid; i < E; i += 256) {
    float v = pa[i] + pr[i];
    float y = (v - mu) * rs * g[i] + bta[i];
    of[(long long)row * E + i] = y;
    if (ob) ob[(long long)row * E + i] = f2bf(y);
  }
}

// ---------------------------------------------------------------------------
extern "C" void kernel_launch(void* const* d_in, const int* in_sizes, int n_in,
                              void* d_out, int out_size, void* d_ws, size_t ws_size,
                              hipStream_t stream) {
  (void)in_sizes; (void)n_in; (void)out_size; (void)ws_size;
  const float* pl   = (const float*)d_in[0];
  const float* sam  = (const float*)d_in[1];
  const float* Wq   = (const float*)d_in[2];
  const float* Wk   = (const float*)d_in[3];
  const float* Wv   = (const float*)d_in[4];
  const float* ln1g = (const float*)d_in[5];
  const float* ln1b = (const float*)d_in[6];
  const float* W1   = (const float*)d_in[7];
  const float* W2   = (const float*)d_in[8];
  const float* ln2g = (const float*)d_in[9];
  const float* ln2b = (const float*)d_in[10];
  float* out = (float*)d_out;

  const long long NE = (long long)BB * SS * EE;  // 16,777,216

  // workspace layout (bytes, 256-aligned; regions reused by lifetime)
  char* ws = (char*)d_ws;
  size_t off = 0;
  auto take = [&](size_t bytes) { size_t r = off; off = (off + bytes + 255) & ~(size_t)255; return r; };
  size_t o_wqb  = take(EE * EE * 2);     // Wq^T  bf16 [E][E]
  size_t o_wkb  = take(EE * EE * 2);
  size_t o_wvb  = take(EE * EE * 2);
  size_t o_w1b  = take(EE * HH * 2);     // W1^T  bf16 [H][E]
  size_t o_w2b  = take(HH * EE * 2);     // W2^T  bf16 [E][H]
  size_t o_plb  = take(NE * 2);          // later reused (with samb) for hb
  size_t o_samb = take(NE * 2);
  size_t o_qb   = take(NE * 2);          // later reused for wb, then xb
  size_t o_kb   = take(NE * 2);          // later reused (with vb) for ff (f32)
  size_t o_vb   = take(NE * 2);
  size_t o_scf  = take(NE * 4);          // scores f32; later reused for xf
  size_t o_attn = take(NE * 4);
  size_t o_pmax = take(BB * 1024 * 4);
  size_t o_psum = take(BB * 1024 * 4);
  size_t o_gmax = take(BB * 4);
  size_t o_gsum = take(BB * 4);

  unsigned short* plb  = (unsigned short*)(ws + o_plb);
  unsigned short* samb = (unsigned short*)(ws + o_samb);
  unsigned short* wqbT = (unsigned short*)(ws + o_wqb);
  unsigned short* wkbT = (unsigned short*)(ws + o_wkb);
  unsigned short* wvbT = (unsigned short*)(ws + o_wvb);
  unsigned short* w1bT = (unsigned short*)(ws + o_w1b);
  unsigned short* w2bT = (unsigned short*)(ws + o_w2b);
  unsigned short* qb   = (unsigned short*)(ws + o_qb);
  unsigned short* kb   = (unsigned short*)(ws + o_kb);
  unsigned short* vb   = (unsigned short*)(ws + o_vb);
  float*          scf  = (float*)(ws + o_scf);
  unsigned short* wb   = (unsigned short*)(ws + o_qb);   // reuse qb
  float*          attn = (float*)(ws + o_attn);
  float*          xf   = (float*)(ws + o_scf);           // reuse scores
  unsigned short* xb   = (unsigned short*)(ws + o_qb);   // reuse qb/wb
  unsigned short* hb   = (unsigned short*)(ws + o_plb);  // reuse plb+samb (64MB)
  float*          fff  = (float*)(ws + o_kb);            // reuse kb+vb (64MB)
  float*          pmax = (float*)(ws + o_pmax);
  float*          psum = (float*)(ws + o_psum);
  float*          gmax = (float*)(ws + o_gmax);
  float*          gsum = (float*)(ws + o_gsum);

  dim3 t256(256);

  // 1) bf16 conversions (activations plain, weights transposed)
  k_f32_to_bf16<<<dim3((unsigned)(NE / 256)), t256, 0, stream>>>(pl, plb, NE);
  k_f32_to_bf16<<<dim3((unsigned)(NE / 256)), t256, 0, stream>>>(sam, samb, NE);
  k_f32_to_bf16_T<<<dim3(EE / 32, EE / 32), t256, 0, stream>>>(Wq, wqbT, EE, EE);
  k_f32_to_bf16_T<<<dim3(EE / 32, EE / 32), t256, 0, stream>>>(Wk, wkbT, EE, EE);
  k_f32_to_bf16_T<<<dim3(EE / 32, EE / 32), t256, 0, stream>>>(Wv, wvbT, EE, EE);
  k_f32_to_bf16_T<<<dim3(HH / 32, EE / 32), t256, 0, stream>>>(W1, w1bT, EE, HH);
  k_f32_to_bf16_T<<<dim3(EE / 32, HH / 32), t256, 0, stream>>>(W2, w2bT, HH, EE);

  // 2) projections q,k,v : [BS,E] x [E,E]^T-layout -> bf16
  dim3 gproj(EE / 64, (BB * SS) / 128, 1);
  k_gemm_bf16_wmma<1, 1><<<gproj, t256, 0, stream>>>(plb,  wqbT, qb, EE, EE, EE, EE,
                                                     0, 0, 0, 1.0f);
  k_gemm_bf16_wmma<1, 1><<<gproj, t256, 0, stream>>>(samb, wkbT, kb, EE, EE, EE, EE,
                                                     0, 0, 0, 1.0f);
  k_gemm_bf16_wmma<1, 1><<<gproj, t256, 0, stream>>>(samb, wvbT, vb, EE, EE, EE, EE,
                                                     0, 0, 0, 1.0f);

  // 3) scores = q k^T / sqrt(E)  (batched, B=[N,K] natural, f32 out)
  dim3 gsc(SS / 64, SS / 128, BB);
  k_gemm_bf16_wmma<1, 0><<<gsc, t256, 0, stream>>>(qb, kb, scf, EE, EE, EE, SS,
                                                   (long long)SS * EE, (long long)SS * EE,
                                                   (long long)SS * SS, 0.03125f);

  // 4) flattened softmax over S*S per batch
  int per_batch = SS * SS, chunk = 1024, nblk = per_batch / chunk;
  k_part_max<<<dim3(nblk, BB), t256, 0, stream>>>(scf, pmax, per_batch, chunk);
  k_reduce_final<<<dim3(BB), t256, 0, stream>>>(pmax, gmax, nblk, 1);
  k_part_expsum<<<dim3(nblk, BB), t256, 0, stream>>>(scf, gmax, psum, per_batch, chunk);
  k_reduce_final<<<dim3(BB), t256, 0, stream>>>(psum, gsum, nblk, 0);
  k_softmax_norm<<<dim3(per_batch / 256, BB), t256, 0, stream>>>(scf, gmax, gsum, wb, per_batch);

  // 5) attn = w v  (batched, B=[K,N], f32 out)
  dim3 gat(EE / 64, SS / 128, BB);
  k_gemm_bf16_wmma<0, 0><<<gat, t256, 0, stream>>>(wb, vb, attn, SS, SS, EE, EE,
                                                   (long long)SS * SS, (long long)SS * EE,
                                                   (long long)SS * EE, 1.0f);

  // 6) x = LN(attn + pl) -> xf (f32) and xb (bf16)
  k_ln_fused<<<dim3(BB * SS), t256, 0, stream>>>(attn, pl, ln1g, ln1b, xf, xb, EE);

  // 7) h = x W1 : [BS,E] x W1^T[H,E] -> bf16
  dim3 gh(HH / 64, (BB * SS) / 128, 1);
  k_gemm_bf16_wmma<1, 1><<<gh, t256, 0, stream>>>(xb, w1bT, hb, EE, EE, EE, HH,
                                                  0, 0, 0, 1.0f);

  // 8) ff = h W2 : [BS,2E] x W2^T[E,2E] -> f32
  dim3 gf(EE / 64, (BB * SS) / 128, 1);
  k_gemm_bf16_wmma<1, 0><<<gf, t256, 0, stream>>>(hb, w2bT, fff, HH, HH, HH, EE,
                                                  0, 0, 0, 1.0f);

  // 9) out = LN(ff + x)
  k_ln_fused<<<dim3(BB * SS), t256, 0, stream>>>(fff, xf, ln2g, ln2b, out, nullptr, EE);
}
